// LlamaBlock_9500467659039
// MI455X (gfx1250) — matrix operand order repaired and checked by reading, not compiled, not run on hardware
//
#include <hip/hip_runtime.h>
#include <cstdint>
#include <cstddef>

// ---------------- types ----------------
typedef __attribute__((ext_vector_type(16))) __bf16 bf16x16;
typedef __attribute__((ext_vector_type(8)))  float  f32x8;
typedef __attribute__((ext_vector_type(8)))  unsigned short u16x8;
typedef __attribute__((ext_vector_type(16))) unsigned short u16x16;

static constexpr int kB    = 4;
static constexpr int kT    = 1024;
static constexpr int kC    = 2048;
static constexpr int kH    = 16;
static constexpr int kHD   = 128;
static constexpr int kHID  = 5464;
static constexpr int kHIDP = 5472;   // padded to a multiple of 32 for the down-proj K dim
static constexpr int kM    = kB * kT; // 4096 activation rows

// ---------------- scalar helpers ----------------
__device__ __forceinline__ unsigned short f2bf(float f) {
  unsigned int u = __builtin_bit_cast(unsigned int, f);
  u += 0x7fffu + ((u >> 16) & 1u);           // round-to-nearest-even
  return (unsigned short)(u >> 16);
}
__device__ __forceinline__ float bf2f(unsigned short h) {
  return __builtin_bit_cast(float, ((unsigned int)h) << 16);
}

__device__ __forceinline__ f32x8 wmma_bf16(bf16x16 a, bf16x16 b, f32x8 c) {
  // D = A(16x32 bf16) * B(32x16 bf16) + C(16x16 f32)
  return __builtin_amdgcn_wmma_f32_16x16x32_bf16(false, a, false, b, (short)0, c, false, false);
}

// ---- CDNA5 async global->LDS copy (ASYNCcnt path, ISA 15.18.3) ----
// VDST VGPR holds the wave-relative LDS byte offset; for LDS-aperture generic
// pointers the low 32 bits ARE the LDS offset (ISA 10.2 aperture mapping).
__device__ __forceinline__ void async_copy_b128(void* lds_dst, const void* gsrc) {
  const unsigned lds_off = (unsigned)(uintptr_t)lds_dst;
  const unsigned long long ga = (unsigned long long)(uintptr_t)gsrc;
  asm volatile("global_load_async_to_lds_b128 %0, %1, off"
               :: "v"(lds_off), "v"(ga)
               : "memory");
}
__device__ __forceinline__ void wait_async0() {
  asm volatile("s_wait_asynccnt 0x0" ::: "memory");
}

// A fragment: lane m = lane&15; K chunks [kb+base .. +7] and [kb+base+16 .. +23], base = lane<16?0:8
__device__ __forceinline__ bf16x16 lds_frag_A(const unsigned short* base, int stride,
                                              int row0, int kb, int lane) {
  const int m  = row0 + (lane & 15);
  const int k0 = kb + ((lane < 16) ? 0 : 8);
  const unsigned short* p = base + m * stride + k0;
  union { u16x8 h[2]; u16x16 v; } u;
  u.h[0] = *(const u16x8*)(p);
  u.h[1] = *(const u16x8*)(p + 16);
  return __builtin_bit_cast(bf16x16, u.v);
}
// B fragment (from [N][K]-major tile): lane n = lane&15; K chunks [kb+base .. +7] and [kb+base+16 .. +23], base = lane<16?0:8
__device__ __forceinline__ bf16x16 lds_frag_B(const unsigned short* base, int stride,
                                              int row0, int kb, int lane) {
  const int n  = row0 + (lane & 15);
  const int k0 = kb + ((lane < 16) ? 0 : 8);
  const unsigned short* p = base + n * stride + k0;
  union { u16x8 h[2]; u16x16 v; } u;
  u.h[0] = *(const u16x8*)(p);
  u.h[1] = *(const u16x8*)(p + 16);
  return __builtin_bit_cast(bf16x16, u.v);
}

// ---------------- octonion tables (mirrors reference _oct_table) ----------------
__device__ __forceinline__ void oct_tables(int idx[8][8], float sgn[8][8]) {
  #pragma unroll
  for (int j = 0; j < 8; ++j) { idx[0][j] = j; sgn[0][j] = 1.f; }
  #pragma unroll
  for (int i = 1; i < 8; ++i) { idx[i][0] = i; sgn[i][0] = 1.f; idx[i][i] = 0; sgn[i][i] = -1.f; }
  const int tr[7][3] = {{1,2,3},{1,4,5},{1,7,6},{2,4,6},{2,5,7},{3,4,7},{3,6,5}};
  #pragma unroll
  for (int t = 0; t < 7; ++t) {
    const int a = tr[t][0], b = tr[t][1], c = tr[t][2];
    const int cyc[3][3] = {{a,b,c},{b,c,a},{c,a,b}};
    #pragma unroll
    for (int u = 0; u < 3; ++u) {
      const int p = cyc[u][0], q = cyc[u][1], r = cyc[u][2];
      idx[p][q] = r; sgn[p][q] = 1.f;
      idx[q][p] = r; sgn[q][p] = -1.f;
    }
  }
}

// ---------------- per-component ternary scales: mean |W| over (o,p) ----------------
__global__ __launch_bounds__(256) void oct_scale_kernel(const float* __restrict__ W,
                                                        float* __restrict__ scale_out, int OP) {
  __shared__ float red[256];
  const int i = blockIdx.x;                    // octonion component 0..7
  const float* Wi = W + (size_t)i * OP;
  float s = 0.f;
  for (int c = threadIdx.x; c < OP; c += 256) s += fabsf(Wi[c]);
  red[threadIdx.x] = s; __syncthreads();
  for (int o = 128; o > 0; o >>= 1) {
    if (threadIdx.x < o) red[threadIdx.x] += red[threadIdx.x + o];
    __syncthreads();
  }
  if (threadIdx.x == 0) scale_out[i] = red[0] / (float)OP;
}

// ---------------- expand octonion weight to dense bf16 [N=8*O][Kd] ----------------
__global__ __launch_bounds__(256) void oct_build_kernel(const float* __restrict__ W,
    const float* __restrict__ scales, unsigned short* __restrict__ Wd,
    int O, int P, int Kd) {
  const int total = 8 * O * P;
  const int id = blockIdx.x * 256 + threadIdx.x;
  if (id >= total) return;
  const int p = id % P;
  const int o = (id / P) % O;
  const int i = id / (P * O);
  const float s = scales[i];
  float q = rintf(W[id] / (s + 1e-8f));
  q = fminf(1.f, fmaxf(-1.f, q)) * s;
  int idxT[8][8]; float sgnT[8][8];
  oct_tables(idxT, sgnT);
  #pragma unroll
  for (int j = 0; j < 8; ++j) {
    const int kk = idxT[i][j];                 // e_i * e_j = sgn * e_kk
    Wd[(size_t)(kk * O + o) * Kd + (j * P + p)] = f2bf(sgnT[i][j] * q);
  }
}

__global__ void padzero_kernel(unsigned short* __restrict__ Wd, int N, int Kd, int Kreal) {
  const int pad = Kd - Kreal;
  const int total = N * pad;
  const int id = blockIdx.x * 256 + threadIdx.x;
  if (id >= total) return;
  Wd[(size_t)(id / pad) * Kd + Kreal + (id % pad)] = 0;
}

// ---------------- rmsnorm (f32 in -> bf16 out) ----------------
__global__ __launch_bounds__(256) void rmsnorm_kernel(const float* __restrict__ x,
    const float* __restrict__ w, unsigned short* __restrict__ out) {
  __shared__ float red[256];
  const int row = blockIdx.x;
  const float* xr = x + (size_t)row * kC;
  float s = 0.f;
  for (int c = threadIdx.x; c < kC; c += 256) { const float v = xr[c]; s += v * v; }
  red[threadIdx.x] = s; __syncthreads();
  for (int o = 128; o > 0; o >>= 1) {
    if (threadIdx.x < o) red[threadIdx.x] += red[threadIdx.x + o];
    __syncthreads();
  }
  const float inv = rsqrtf(red[0] / (float)kC + 1e-6f);
  unsigned short* orow = out + (size_t)row * kC;
  for (int c = threadIdx.x; c < kC; c += 256) orow[c] = f2bf(xr[c] * inv * w[c]);
}

// ---------------- bf16 GEMM: out[m][n] = sum_k A[m][k] * Bw[n][k] ----------------
// mode 0: bf16 out; mode 1: f32 out; mode 2: f32 out + f32 residual
__global__ __launch_bounds__(256) void gemm_bf16_kernel(
    const unsigned short* __restrict__ A,    // [kM][K] bf16
    const unsigned short* __restrict__ Bw,   // [N][K]  bf16
    void* __restrict__ Cout,
    const float* __restrict__ resid,
    int N, int K, int mode) {
  __shared__ unsigned short As[128][40];     // 128x32 tile, +8 pad
  __shared__ unsigned short Bs[128][40];
  const int tid  = threadIdx.x;
  const int lane = tid & 31;
  const int wave = tid >> 5;                 // 8 waves
  const int wm = wave & 3;                   // 4x2 wave grid: 32 rows x 64 cols each
  const int wn = wave >> 2;
  const int m0 = blockIdx.y * 128;
  const int n0 = blockIdx.x * 128;

  const f32x8 z = {0.f,0.f,0.f,0.f,0.f,0.f,0.f,0.f};
  f32x8 acc[2][4];
  #pragma unroll
  for (int i = 0; i < 2; ++i)
    #pragma unroll
    for (int j = 0; j < 4; ++j) acc[i][j] = z;

  const int sr = tid >> 1;                   // staging row
  const int sc = (tid & 1) * 16;             // staging col chunk
  int rb = n0 + sr; if (rb >= N) rb = N - 1; // clamp N tail (stores are guarded)

  for (int k0 = 0; k0 < K; k0 += 32) {
    __syncthreads();
    {
      // async global->LDS tile copies (ASYNCcnt), VALU stays free
      const unsigned short* pa = A + (size_t)(m0 + sr) * K + k0 + sc;
      async_copy_b128(&As[sr][sc],     pa);
      async_copy_b128(&As[sr][sc + 8], pa + 8);
      const unsigned short* pb = Bw + (size_t)rb * K + k0 + sc;
      async_copy_b128(&Bs[sr][sc],     pb);
      async_copy_b128(&Bs[sr][sc + 8], pb + 8);
      __builtin_prefetch(pa + 32, 0, 3);     // next k-tile -> global_prefetch
      __builtin_prefetch(pb + 32, 0, 3);
    }
    wait_async0();
    __syncthreads();
    const bf16x16 a0 = lds_frag_A(&As[0][0], 40, wm * 32,      0, lane);
    const bf16x16 a1 = lds_frag_A(&As[0][0], 40, wm * 32 + 16, 0, lane);
    #pragma unroll
    for (int ni = 0; ni < 4; ++ni) {
      const bf16x16 bfr = lds_frag_B(&Bs[0][0], 40, wn * 64 + ni * 16, 0, lane);
      acc[0][ni] = wmma_bf16(a0, bfr, acc[0][ni]);
      acc[1][ni] = wmma_bf16(a1, bfr, acc[1][ni]);
    }
  }

  const int ln = lane & 15;
  const int hb = (lane >> 4) * 8;
  #pragma unroll
  for (int mi = 0; mi < 2; ++mi)
    #pragma unroll
    for (int ni = 0; ni < 4; ++ni)
      #pragma unroll
      for (int v = 0; v < 8; ++v) {
        const int m = m0 + wm * 32 + mi * 16 + hb + v;
        const int n = n0 + wn * 64 + ni * 16 + ln;
        if (n < N) {
          const float val = acc[mi][ni][v];
          if (mode == 0) {
            ((unsigned short*)Cout)[(size_t)m * N + n] = f2bf(val);
          } else {
            float o = val;
            if (mode == 2) o += resid[(size_t)m * N + n];
            ((float*)Cout)[(size_t)m * N + n] = o;
          }
        }
      }
}

// ---------------- RoPE (f32 in -> bf16 out), layout (B,T,H,HD) flattened ----------------
__global__ __launch_bounds__(256) void rope_kernel(const float* __restrict__ xin,
    const float* __restrict__ cosT, const float* __restrict__ sinT,
    unsigned short* __restrict__ out) {
  const int total = kM * kH * (kHD / 2);
  const int id = blockIdx.x * 256 + threadIdx.x;
  if (id >= total) return;
  const int d  = id % (kHD / 2);
  const int hh = (id / (kHD / 2)) % kH;
  const int m  = id / ((kHD / 2) * kH);
  const int t  = m % kT;
  const size_t base = (size_t)m * kC + (size_t)hh * kHD;
  const float xr = xin[base + 2 * d];
  const float xi = xin[base + 2 * d + 1];
  const float c  = cosT[t * (kHD / 2) + d];
  const float s  = sinT[t * (kHD / 2) + d];
  out[base + 2 * d]     = f2bf(xr * c - xi * s);
  out[base + 2 * d + 1] = f2bf(xr * s + xi * c);
}

// ---------------- flash attention: one WG per (b,h,128-row q tile) ----------------
__global__ __launch_bounds__(256) void attn_kernel(
    const unsigned short* __restrict__ Q,
    const unsigned short* __restrict__ Kc,
    const unsigned short* __restrict__ V,
    unsigned short* __restrict__ Y) {
  extern __shared__ char smem[];
  unsigned short* Qs = (unsigned short*)smem;   // [128][136] row-major (t, d)
  unsigned short* Ks = Qs + 128 * 136;          // [128][136] row-major (t, d)
  unsigned short* Vs = Ks + 128 * 136;          // [128][136] TRANSPOSED (d, t)
  unsigned short* Ps = Vs + 128 * 136;          // per-wave P tiles: [8][16][136]

  const int tid  = threadIdx.x;
  const int lane = tid & 31;
  const int wave = tid >> 5;
  const int qt = blockIdx.x;
  const int h  = blockIdx.y;
  const int bz = blockIdx.z;
  const int m0 = qt * 128;
  const size_t headOff = (size_t)bz * kT * kC + (size_t)h * kHD;

  const int sr  = tid >> 1;
  const int sc0 = (tid & 1) * 64;
  { // stage Q tile once (async global->LDS)
    const unsigned short* src = Q + headOff + (size_t)(m0 + sr) * kC + sc0;
    #pragma unroll
    for (int c = 0; c < 64; c += 8)
      async_copy_b128(&Qs[sr * 136 + sc0 + c], src + c);
  }

  const f32x8 z8 = {0.f,0.f,0.f,0.f,0.f,0.f,0.f,0.f};
  f32x8 oacc[8]; float mrow[8], lrow[8];
  #pragma unroll
  for (int i = 0; i < 8; ++i) { oacc[i] = z8; mrow[i] = -1e30f; lrow[i] = 0.f; }

  const int ln = lane & 15;
  const int hb = (lane >> 4) * 8;
  const float scale = 0.088388347648318447f;    // 1/sqrt(128)

  for (int kt = 0; kt <= qt; ++kt) {
    wait_async0();
    __syncthreads();
    { // stage K (row-major, async) and V (transposed, scalar stores)
      const unsigned short* ksrc = Kc + headOff + (size_t)(kt * 128 + sr) * kC + sc0;
      #pragma unroll
      for (int c = 0; c < 64; c += 8)
        async_copy_b128(&Ks[sr * 136 + sc0 + c], ksrc + c);
      const unsigned short* vsrc = V + headOff + (size_t)(kt * 128 + sr) * kC + sc0;
      #pragma unroll
      for (int c = 0; c < 64; ++c)
        Vs[(sc0 + c) * 136 + sr] = vsrc[c];
    }
    wait_async0();
    __syncthreads();

    // S = Q * K^T  (wave owns 16 q-rows x 128 k-cols)
    f32x8 sacc[8];
    #pragma unroll
    for (int i = 0; i < 8; ++i) sacc[i] = z8;
    #pragma unroll
    for (int kc = 0; kc < 4; ++kc) {
      const bf16x16 afr = lds_frag_A(Qs, 136, wave * 16, kc * 32, lane);
      #pragma unroll
      for (int ni = 0; ni < 8; ++ni) {
        const bf16x16 bfr = lds_frag_B(Ks, 136, ni * 16, kc * 32, lane);
        sacc[ni] = wmma_bf16(afr, bfr, sacc[ni]);
      }
    }

    // scale + causal mask
    #pragma unroll
    for (int ni = 0; ni < 8; ++ni)
      #pragma unroll
      for (int v = 0; v < 8; ++v) {
        float s = sacc[ni][v] * scale;
        const int row = m0 + wave * 16 + hb + v;
        const int col = kt * 128 + ni * 16 + ln;
        if (col > row) s = -1e30f;
        sacc[ni][v] = s;
      }

    // online softmax: row stats live in 16-lane halves (C-layout)
    float rmax[8];
    #pragma unroll
    for (int v = 0; v < 8; ++v) {
      float r = sacc[0][v];
      #pragma unroll
      for (int ni = 1; ni < 8; ++ni) r = fmaxf(r, sacc[ni][v]);
      r = fmaxf(r, __shfl_xor(r, 1, 32));
      r = fmaxf(r, __shfl_xor(r, 2, 32));
      r = fmaxf(r, __shfl_xor(r, 4, 32));
      r = fmaxf(r, __shfl_xor(r, 8, 32));
      rmax[v] = r;
    }
    float mnew[8], corr[8], rsum[8];
    #pragma unroll
    for (int v = 0; v < 8; ++v) {
      mnew[v] = fmaxf(mrow[v], rmax[v]);
      corr[v] = expf(mrow[v] - mnew[v]);
      mrow[v] = mnew[v];
      rsum[v] = 0.f;
    }
    #pragma unroll
    for (int ni = 0; ni < 8; ++ni)
      #pragma unroll
      for (int v = 0; v < 8; ++v) {
        const float p = expf(sacc[ni][v] - mnew[v]);
        sacc[ni][v] = p;
        rsum[v] += p;
      }
    #pragma unroll
    for (int v = 0; v < 8; ++v) {
      float r = rsum[v];
      r += __shfl_xor(r, 1, 32);
      r += __shfl_xor(r, 2, 32);
      r += __shfl_xor(r, 4, 32);
      r += __shfl_xor(r, 8, 32);
      lrow[v] = lrow[v] * corr[v] + r;
    }
    #pragma unroll
    for (int ni = 0; ni < 8; ++ni)
      #pragma unroll
      for (int v = 0; v < 8; ++v) oacc[ni][v] *= corr[v];

    // P (C-layout) -> bf16 -> per-wave LDS -> A-fragments   (LDS is in-order per wave)
    unsigned short* Pw = Ps + wave * 16 * 136;
    #pragma unroll
    for (int ni = 0; ni < 8; ++ni)
      #pragma unroll
      for (int v = 0; v < 8; ++v)
        Pw[(hb + v) * 136 + ni * 16 + ln] = f2bf(sacc[ni][v]);
    asm volatile("" ::: "memory");

    // O += P * V
    #pragma unroll
    for (int kc = 0; kc < 4; ++kc) {
      const bf16x16 afr = lds_frag_A(Pw, 136, 0, kc * 32, lane);
      #pragma unroll
      for (int ni = 0; ni < 8; ++ni) {
        const bf16x16 bfr = lds_frag_B(Vs, 136, ni * 16, kc * 32, lane);
        oacc[ni] = wmma_bf16(afr, bfr, oacc[ni]);
      }
    }
  }

  #pragma unroll
  for (int ni = 0; ni < 8; ++ni)
    #pragma unroll
    for (int v = 0; v < 8; ++v) {
      const int row = m0 + wave * 16 + hb + v;
      Y[headOff + (size_t)row * kC + ni * 16 + ln] = f2bf(oacc[ni][v] / lrow[v]);
    }
}

// ---------------- silu(gate) * up -> padded bf16 [kM][kHIDP] ----------------
__global__ __launch_bounds__(256) void silu_mul_kernel(const unsigned short* __restrict__ g,
    const unsigned short* __restrict__ u, unsigned short* __restrict__ out) {
  const int total = kM * kHIDP;
  const int id = blockIdx.x * 256 + threadIdx.x;
  if (id >= total) return;
  const int c = id % kHIDP;
  const int m = id / kHIDP;
  unsigned short o = 0;
  if (c < kHID) {
    const float gv = bf2f(g[(size_t)m * kHID + c]);
    const float uv = bf2f(u[(size_t)m * kHID + c]);
    o = f2bf((gv / (1.f + expf(-gv))) * uv);
  }
  out[(size_t)m * kHIDP + c] = o;
}

// ---------------- host orchestration ----------------
extern "C" void kernel_launch(void* const* d_in, const int* in_sizes, int n_in,
                              void* d_out, int out_size, void* d_ws, size_t ws_size,
                              hipStream_t stream) {
  (void)in_sizes; (void)n_in; (void)out_size; (void)ws_size;
  const float* x    = (const float*)d_in[0];
  const float* fcos = (const float*)d_in[1];
  const float* fsin = (const float*)d_in[2];
  const float* anw  = (const float*)d_in[3];
  const float* fnw  = (const float*)d_in[4];
  const float* wPtr[7] = { (const float*)d_in[5], (const float*)d_in[6], (const float*)d_in[7],
                           (const float*)d_in[8], (const float*)d_in[9], (const float*)d_in[10],
                           (const float*)d_in[11] };
  // wq, wk, wv, wo, w_gate, w_up, w_down
  const int wO[7]  = {256, 256, 256, 256, 683, 683, 256};
  const int wP[7]  = {256, 256, 256, 256, 256, 256, 683};
  const int wKd[7] = {kC,  kC,  kC,  kC,  kC,  kC,  kHIDP};

  char* ws = (char*)d_ws;
  size_t off = 0;
  auto bump = [&](size_t bytes) -> void* {
    void* p = ws + off;
    off += (bytes + 255) & ~(size_t)255;
    return p;
  };

  float* scales = (float*)bump(7 * 8 * sizeof(float));
  unsigned short* Wd[7];
  for (int w = 0; w < 7; ++w)
    Wd[w] = (unsigned short*)bump((size_t)(8 * wO[w]) * wKd[w] * sizeof(unsigned short));

  unsigned short* xn  = (unsigned short*)bump((size_t)kM * kC * 2);
  float*          qf  = (float*)bump((size_t)kM * kC * 4);
  float*          kf  = (float*)bump((size_t)kM * kC * 4);
  unsigned short* qb  = (unsigned short*)bump((size_t)kM * kC * 2);
  unsigned short* kb  = (unsigned short*)bump((size_t)kM * kC * 2);
  unsigned short* vb  = (unsigned short*)bump((size_t)kM * kC * 2);
  unsigned short* yb  = (unsigned short*)bump((size_t)kM * kC * 2);
  float*          hbf = (float*)bump((size_t)kM * kC * 4);
  unsigned short* hnb = (unsigned short*)bump((size_t)kM * kC * 2);
  unsigned short* gb  = (unsigned short*)bump((size_t)kM * kHID * 2);
  unsigned short* ub  = (unsigned short*)bump((size_t)kM * kHID * 2);
  unsigned short* ffb = (unsigned short*)bump((size_t)kM * kHIDP * 2);

  // 1) ternary-quantize + expand octonion weights to dense bf16 [N][K]
  for (int w = 0; w < 7; ++w) {
    oct_scale_kernel<<<8, 256, 0, stream>>>(wPtr[w], scales + 8 * w, wO[w] * wP[w]);
    const int total = 8 * wO[w] * wP[w];
    oct_build_kernel<<<(total + 255) / 256, 256, 0, stream>>>(
        wPtr[w], scales + 8 * w, Wd[w], wO[w], wP[w], wKd[w]);
  }
  { // zero the K padding of the down projection
    const int total = kC * (kHIDP - kHID);
    padzero_kernel<<<(total + 255) / 256, 256, 0, stream>>>(Wd[6], kC, kHIDP, kHID);
  }

  // 2) attn rmsnorm
  rmsnorm_kernel<<<kM, 256, 0, stream>>>(x, anw, xn);

  const dim3 g_cc(kC / 128, kM / 128);             // 16 x 32
  const dim3 g_ch((kHID + 127) / 128, kM / 128);   // 43 x 32

  // 3) q,k,v projections
  gemm_bf16_kernel<<<g_cc, 256, 0, stream>>>(xn, Wd[0], qf, nullptr, kC, kC, 1);
  gemm_bf16_kernel<<<g_cc, 256, 0, stream>>>(xn, Wd[1], kf, nullptr, kC, kC, 1);
  gemm_bf16_kernel<<<g_cc, 256, 0, stream>>>(xn, Wd[2], vb, nullptr, kC, kC, 0);

  // 4) RoPE
  const int ropeTotal = kM * kH * (kHD / 2);
  rope_kernel<<<(ropeTotal + 255) / 256, 256, 0, stream>>>(qf, fcos, fsin, qb);
  rope_kernel<<<(ropeTotal + 255) / 256, 256, 0, stream>>>(kf, fcos, fsin, kb);

  // 5) flash attention (dynamic LDS: Q + K + V + P tiles = 139264 B < 320 KB/WGP)
  attn_kernel<<<dim3(kT / 128, kH, kB), 256, 139264, stream>>>(qb, kb, vb, yb);

  // 6) output projection + residual -> h (f32)
  gemm_bf16_kernel<<<g_cc, 256, 0, stream>>>(yb, Wd[3], hbf, x, kC, kC, 2);

  // 7) ffn rmsnorm
  rmsnorm_kernel<<<kM, 256, 0, stream>>>(hbf, fnw, hnb);

  // 8) gate / up
  gemm_bf16_kernel<<<g_ch, 256, 0, stream>>>(hnb, Wd[4], gb, nullptr, kHID, kC, 0);
  gemm_bf16_kernel<<<g_ch, 256, 0, stream>>>(hnb, Wd[5], ub, nullptr, kHID, kC, 0);

  // 9) silu(gate)*up (padded to 5472 K)
  const int siluTotal = kM * kHIDP;
  silu_mul_kernel<<<(siluTotal + 255) / 256, 256, 0, stream>>>(gb, ub, ffb);

  // 10) down projection + residual h -> final output (f32)
  gemm_bf16_kernel<<<g_cc, 256, 0, stream>>>(ffb, Wd[6], (float*)d_out, hbf, kC, kHIDP, 2);
}
